// MultiCON_47931835023372
// MI455X (gfx1250) — matrix operand order repaired and checked by reading, not compiled, not run on hardware
//
#include <hip/hip_runtime.h>

// ---------------------------------------------------------------------------
// CDNA5 (gfx1250, wave32) implementation of the DGCNN MultiCON block.
// All channel-mixing GEMMs run on v_wmma_f32_16x16x32_bf16 (bf16 A/B, f32 acc).
// Final GEMM stages its B-tiles via GLOBAL_LOAD_ASYNC_TO_LDS_B128 (ASYNCcnt),
// double-buffered. Top-16 kNN is a register insertion sort over LDS tiles.
// ---------------------------------------------------------------------------

typedef __attribute__((ext_vector_type(16))) __bf16 v16bf;
typedef __attribute__((ext_vector_type(8)))  __bf16 v8bf;
typedef __attribute__((ext_vector_type(8)))  float  v8f;

#define DEV __device__ __forceinline__

DEV v8f wmma_bf16(v16bf a, v16bf b, v8f c) {
  return __builtin_amdgcn_wmma_f32_16x16x32_bf16(false, a, false, b, (short)0, c,
                                                 false, false);
}

// Async global -> LDS copy (16B per lane), tracked by ASYNCcnt.
DEV void async_load_b128(const __bf16* g, unsigned lds_off) {
  asm volatile("global_load_async_to_lds_b128 %0, %1, off"
               :: "v"(lds_off), "v"(g)
               : "memory");
}
DEV void wait_asynccnt0() {
  asm volatile("s_wait_asynccnt 0x0" ::: "memory");
}

// A-fragment (16x32, M x K) from row-major bf16 weights W[O][kdim].
// lane row m = m0 + (lane&15); elems 0..7 -> K = k0 + (lane&16?8:0) + e,
// elems 8..15 -> K = k0 + 16 + (lane&16?8:0) + (e-8).
DEV v16bf load_frag_A(const __bf16* __restrict__ w, int kdim, int m0, int k0) {
  int lane = threadIdx.x & 31;
  const __bf16* row = w + (size_t)(m0 + (lane & 15)) * kdim + k0 + ((lane & 16) ? 8 : 0);
  v8bf a = *(const v8bf*)(row);
  v8bf b = *(const v8bf*)(row + 16);
  v16bf r;
#pragma unroll
  for (int i = 0; i < 8; ++i) { r[i] = a[i]; r[i + 8] = b[i]; }
  return r;
}

// B-fragment (32x16, K x N) from position-major bf16 activations F[pos][stride].
// lane col = col0 + (lane&15); elems 0..7 -> K = k0 + (lane&16?8:0) + e,
// elems 8..15 -> K = k0 + 16 + (lane&16?8:0) + (e-8).
DEV v16bf load_frag_B(const __bf16* __restrict__ f, int stride, int col0, int k0) {
  int lane = threadIdx.x & 31;
  const __bf16* p = f + (size_t)(col0 + (lane & 15)) * stride + k0 + ((lane & 16) ? 8 : 0);
  v8bf a = *(const v8bf*)(p);
  v8bf b = *(const v8bf*)(p + 16);
  v16bf r;
#pragma unroll
  for (int i = 0; i < 8; ++i) { r[i] = a[i]; r[i + 8] = b[i]; }
  return r;
}

// Epilogue: per-row BN scale/shift + ReLU, pack 8 bf16, single b128 store into
// position-major buffer. Lane holds rows m0+rb..m0+rb+7, col0+(lane&15).
DEV void store_tile(__bf16* dst, int stride, int col0, int m0, v8f acc, v8f sc, v8f sh) {
  int lane = threadIdx.x & 31;
  int col = col0 + (lane & 15);
  int rb = (lane & 16) ? 8 : 0;
  v8bf o;
#pragma unroll
  for (int r = 0; r < 8; ++r) {
    float v = acc[r] * sc[r] + sh[r];
    v = v > 0.f ? v : 0.f;
    o[r] = (__bf16)v;
  }
  *(v8bf*)(dst + (size_t)col * stride + m0 + rb) = o;
}

// ---------------------------------------------------------------------------
// prep kernels: fold BN into scale/shift, cast weights to K-padded bf16
// ---------------------------------------------------------------------------
__global__ void prep_bn_kernel(const float* __restrict__ g, const float* __restrict__ b,
                               const float* __restrict__ m, const float* __restrict__ v,
                               float* __restrict__ scale, float* __restrict__ shift, int C) {
  int i = blockIdx.x * blockDim.x + threadIdx.x;
  if (i < C) {
    float s = g[i] * rsqrtf(v[i] + 1e-5f);
    scale[i] = s;
    shift[i] = b[i] - m[i] * s;
  }
}

__global__ void prep_w_kernel(const float* __restrict__ src, __bf16* __restrict__ dst,
                              int O, int Cin, int Kpad) {
  int i = blockIdx.x * blockDim.x + threadIdx.x;
  if (i < O * Kpad) {
    int o = i / Kpad, k = i % Kpad;
    float v = (k < Cin) ? src[o * Cin + k] : 0.f;
    dst[i] = (__bf16)v;
  }
}

// ---------------------------------------------------------------------------
// lift: h[b][n][8] = relu(bn(liner_w @ x)), bf16, plus squared norms
// ---------------------------------------------------------------------------
__global__ void lift_kernel(const float* __restrict__ x, const float* __restrict__ lw,
                            const float* __restrict__ ls, const float* __restrict__ lh,
                            __bf16* __restrict__ h, float* __restrict__ sq, int B, int N) {
  int i = blockIdx.x * blockDim.x + threadIdx.x;
  if (i >= B * N) return;
  int b = i / N, n = i - b * N;
  float x0 = x[((size_t)b * 3 + 0) * N + n];
  float x1 = x[((size_t)b * 3 + 1) * N + n];
  float x2 = x[((size_t)b * 3 + 2) * N + n];
  v8bf o;
  float s = 0.f;
#pragma unroll
  for (int c = 0; c < 8; ++c) {
    float v = lw[c * 3 + 0] * x0 + lw[c * 3 + 1] * x1 + lw[c * 3 + 2] * x2;
    v = v * ls[c] + lh[c];
    v = v > 0.f ? v : 0.f;
    o[c] = (__bf16)v;
    s += v * v;
  }
  *(v8bf*)(h + (size_t)i * 8) = o;
  sq[i] = s;
}

// ---------------------------------------------------------------------------
// top-16 neighbors by -||xi-xj||^2 (ranked by 2*<xi,xj> - ||xj||^2).
// Both branches (k=16 and k=12) share this: sorted top-12 = prefix of top-16.
// ---------------------------------------------------------------------------
__global__ void topk_kernel(const __bf16* __restrict__ h, const float* __restrict__ sq,
                            int* __restrict__ topk, int N) {
  __shared__ __bf16 tH[256 * 8];
  __shared__ float tS[256];
  int nblk = N / 256;
  int b = blockIdx.x / nblk;
  int n = (blockIdx.x - b * nblk) * 256 + threadIdx.x;

  float me[8];
  {
    v8bf mv = *(const v8bf*)(h + ((size_t)b * N + n) * 8);
#pragma unroll
    for (int c = 0; c < 8; ++c) me[c] = (float)mv[c];
  }
  float tv[16];
  int ti[16];
#pragma unroll
  for (int s = 0; s < 16; ++s) { tv[s] = -3.4e38f; ti[s] = 0; }

  for (int mb = 0; mb < N; mb += 256) {
    __syncthreads();
    *(v8bf*)(tH + threadIdx.x * 8) =
        *(const v8bf*)(h + ((size_t)b * N + mb + threadIdx.x) * 8);
    tS[threadIdx.x] = sq[(size_t)b * N + mb + threadIdx.x];
    __syncthreads();
    // prefetch next candidate tile while we grind this one (global_prefetch_b8)
    if (mb + 256 < N)
      __builtin_prefetch(h + ((size_t)b * N + mb + 256 + threadIdx.x) * 8, 0, 0);
    for (int t = 0; t < 256; ++t) {
      const __bf16* hm = tH + t * 8;
      float d = 0.f;
#pragma unroll
      for (int c = 0; c < 8; ++c) d += me[c] * (float)hm[c];
      d = 2.f * d - tS[t];
      if (d > tv[15]) {
        float cv = d;
        int ci = mb + t;
#pragma unroll
        for (int s = 0; s < 16; ++s) {
          if (cv > tv[s]) {
            float tf = tv[s]; tv[s] = cv; cv = tf;
            int tii = ti[s]; ti[s] = ci; ci = tii;
          }
        }
      }
    }
  }
#pragma unroll
  for (int s = 0; s < 16; ++s) topk[((size_t)b * N + n) * 16 + s] = ti[s];
}

// ---------------------------------------------------------------------------
// Fused edge-conv branch: per block = 16 points, 4 waves (128 threads).
// gather -> conv1 (WMMA) -> conv2 (WMMA) -> attention score (WMMA) + weighted
// pool (register acc) -> 256-ch projection (WMMA) -> cat[b][n][chbase+...].
// Positions ordered neighbor-major: pos = j*16 + p  (p = point, WMMA column).
// ---------------------------------------------------------------------------
template <int KNB>
__launch_bounds__(128, 1) __global__
void edgeconv_kernel(const __bf16* __restrict__ h, const int* __restrict__ topk,
                     const __bf16* __restrict__ w1,   // [64][32]
                     const __bf16* __restrict__ w2,   // [128][64]
                     const __bf16* __restrict__ cw,   // [128][128]
                     const __bf16* __restrict__ c2w,  // [256][128]
                     const float* __restrict__ s1, const float* __restrict__ h1,
                     const float* __restrict__ s2, const float* __restrict__ h2,
                     const float* __restrict__ sa, const float* __restrict__ ha,
                     __bf16* __restrict__ cat, int N, int chbase) {
  constexpr int P = 16 * KNB;
  extern __shared__ char smem[];
  __bf16* Fe = (__bf16*)smem;        // [P][32]  edge features (K-padded)
  __bf16* F1 = Fe + (size_t)P * 32;  // [P][64]  after conv1
  __bf16* F2 = F1 + (size_t)P * 64;  // [P][128] after conv2
  __bf16* SB = F2 + (size_t)P * 128; // [16][128] pooled s (bf16)

  int nblk = N / 16;
  int b = blockIdx.x / nblk;
  int n0 = (blockIdx.x - b * nblk) * 16;
  int tid = threadIdx.x;
  int wave = tid >> 5;
  int rb = (tid & 16) ? 8 : 0;

  // ---- stage 0: gather edge features [nbr-ctr | ctr | 0...] ----
  for (int pos = tid; pos < P; pos += 128) {
    int jj = pos >> 4;
    int p = pos & 15;
    int n = n0 + p;
    int nb = topk[((size_t)b * N + n) * 16 + jj];
    v8bf nv = *(const v8bf*)(h + ((size_t)b * N + nb) * 8);
    v8bf cv = *(const v8bf*)(h + ((size_t)b * N + n) * 8);
    v8bf d, z = {};
#pragma unroll
    for (int c = 0; c < 8; ++c) d[c] = (__bf16)((float)nv[c] - (float)cv[c]);
    __bf16* e = Fe + (size_t)pos * 32;
    *(v8bf*)(e) = d;
    *(v8bf*)(e + 8) = cv;
    *(v8bf*)(e + 16) = z;
    *(v8bf*)(e + 24) = z;
  }
  __syncthreads();

  // ---- stage 1: conv1 64x16 (K padded to 32), wave w owns rows w*16..+15 ----
  {
    int m0 = wave * 16;
    v16bf A = load_frag_A(w1, 32, m0, 0);
    v8f sc, sh;
#pragma unroll
    for (int r = 0; r < 8; ++r) { sc[r] = s1[m0 + rb + r]; sh[r] = h1[m0 + rb + r]; }
    for (int ct = 0; ct < KNB; ++ct) {
      v8f acc = {};
      acc = wmma_bf16(A, load_frag_B(Fe, 32, ct * 16, 0), acc);
      store_tile(F1, 64, ct * 16, m0, acc, sc, sh);
    }
  }
  __syncthreads();

  // ---- stage 2: conv2 128x64 ----
  for (int mt = wave; mt < 8; mt += 4) {
    int m0 = mt * 16;
    v16bf A0 = load_frag_A(w2, 64, m0, 0);
    v16bf A1 = load_frag_A(w2, 64, m0, 32);
    v8f sc, sh;
#pragma unroll
    for (int r = 0; r < 8; ++r) { sc[r] = s2[m0 + rb + r]; sh[r] = h2[m0 + rb + r]; }
    for (int ct = 0; ct < KNB; ++ct) {
      v8f acc = {};
      acc = wmma_bf16(A0, load_frag_B(F1, 64, ct * 16, 0), acc);
      acc = wmma_bf16(A1, load_frag_B(F1, 64, ct * 16, 32), acc);
      store_tile(F2, 128, ct * 16, m0, acc, sc, sh);
    }
  }
  __syncthreads();

  // ---- stage 3: attention score (128x128 WMMA) + weighted sum over k.
  // Each WMMA column tile = one neighbor j (16 points wide) => per-lane
  // register accumulation of s[channel][point] across neighbors, no conflicts.
  for (int mt = wave; mt < 8; mt += 4) {
    int m0 = mt * 16;
    v16bf A0 = load_frag_A(cw, 128, m0, 0);
    v16bf A1 = load_frag_A(cw, 128, m0, 32);
    v16bf A2 = load_frag_A(cw, 128, m0, 64);
    v16bf A3 = load_frag_A(cw, 128, m0, 96);
    v8f sacc = {};
    for (int j = 0; j < KNB; ++j) {
      v8f acc = {};
      acc = wmma_bf16(A0, load_frag_B(F2, 128, j * 16, 0), acc);
      acc = wmma_bf16(A1, load_frag_B(F2, 128, j * 16, 32), acc);
      acc = wmma_bf16(A2, load_frag_B(F2, 128, j * 16, 64), acc);
      acc = wmma_bf16(A3, load_frag_B(F2, 128, j * 16, 96), acc);
      int col = j * 16 + (tid & 15);
      v8bf fv = *(const v8bf*)(F2 + (size_t)col * 128 + m0 + rb);
#pragma unroll
      for (int r = 0; r < 8; ++r) {
        float sg = 1.f / (1.f + __expf(-acc[r]));
        sacc[r] += (float)fv[r] * sg;
      }
    }
    v8bf o;
#pragma unroll
    for (int r = 0; r < 8; ++r) o[r] = (__bf16)sacc[r];
    *(v8bf*)(SB + (size_t)(tid & 15) * 128 + m0 + rb) = o;
  }
  __syncthreads();

  // ---- stage 4: projection 256x128 + BN + ReLU -> cat[b][n][chbase+ch] ----
  for (int mt = wave * 4; mt < wave * 4 + 4; ++mt) {
    int m0 = mt * 16;
    v8f acc = {};
#pragma unroll
    for (int kt = 0; kt < 4; ++kt) {
      v16bf A = load_frag_A(c2w, 128, m0, kt * 32);
      acc = wmma_bf16(A, load_frag_B(SB, 128, 0, kt * 32), acc);
    }
    v8bf o;
#pragma unroll
    for (int r = 0; r < 8; ++r) {
      float v = acc[r] * sa[m0 + rb + r] + ha[m0 + rb + r];
      v = v > 0.f ? v : 0.f;
      o[r] = (__bf16)v;
    }
    int p = tid & 15;
    *(v8bf*)(cat + ((size_t)b * N + n0 + p) * 512 + chbase + m0 + rb) = o;
  }
}

// ---------------------------------------------------------------------------
// Final conv: out[512, B*N] = relu(bn(W[512,512] @ cat)), f32 output.
// Block = 64x64 output tile, 4 waves. B-tiles (64 cols x 32 K bf16 = 4 KB) are
// staged into LDS once per block via GLOBAL_LOAD_ASYNC_TO_LDS_B128 with double
// buffering (ASYNCcnt), removing the 4x-redundant per-wave global B loads.
// ---------------------------------------------------------------------------
DEV void issue_tile_loads(const __bf16* __restrict__ cat, int pos0, int kt,
                          __bf16* buf) {
  // 64 cols x 32 K bf16, LDS layout [col][32]. 128 threads x 2 x 16B = 4 KB.
  int t = threadIdx.x;
  int col = t >> 1;
  int eo = (t & 1) * 16;
  const __bf16* g = cat + ((size_t)(pos0 + col) * 512 + kt * 32 + eo);
  unsigned l = (unsigned)(size_t)(buf + col * 32 + eo);
  async_load_b128(g, l);
  async_load_b128(g + 8, l + 16);
}

__launch_bounds__(128, 1) __global__
void final_conv_kernel(const __bf16* __restrict__ cat, const __bf16* __restrict__ w,
                       const float* __restrict__ sc, const float* __restrict__ sh,
                       float* __restrict__ out, int N) {
  __shared__ __align__(16) __bf16 Bt[2][64 * 32];
  int pos0 = blockIdx.x * 64;
  int m0 = blockIdx.y * 64 + (threadIdx.x >> 5) * 16;
  int lane = threadIdx.x & 31;
  int rb = (lane & 16) ? 8 : 0;

  v8f acc[4] = {};
  issue_tile_loads(cat, pos0, 0, Bt[0]);
  for (int kt = 0; kt < 16; ++kt) {
    // Drain own async loads, then barrier: guarantees (a) Bt[kt&1] fully
    // written by all waves, (b) all waves done reading Bt[(kt+1)&1] from kt-1.
    wait_asynccnt0();
    __syncthreads();
    if (kt + 1 < 16) issue_tile_loads(cat, pos0, kt + 1, Bt[(kt + 1) & 1]);
    v16bf A = load_frag_A(w, 512, m0, kt * 32);
#pragma unroll
    for (int ct = 0; ct < 4; ++ct)
      acc[ct] = wmma_bf16(A, load_frag_B(Bt[kt & 1], 32, ct * 16, 0), acc[ct]);
  }

  v8f scv, shv;
#pragma unroll
  for (int r = 0; r < 8; ++r) { scv[r] = sc[m0 + rb + r]; shv[r] = sh[m0 + rb + r]; }
#pragma unroll
  for (int ct = 0; ct < 4; ++ct) {
    int pos = pos0 + ct * 16 + (lane & 15);
    int b = pos / N;
    int n = pos - b * N;
#pragma unroll
    for (int r = 0; r < 8; ++r) {
      float v = acc[ct][r] * scv[r] + shv[r];
      v = v > 0.f ? v : 0.f;
      out[((size_t)b * 512 + m0 + rb + r) * N + n] = v;
    }
  }
}

// ---------------------------------------------------------------------------
// host
// ---------------------------------------------------------------------------
extern "C" void kernel_launch(void* const* d_in, const int* in_sizes, int n_in,
                              void* d_out, int out_size, void* d_ws, size_t ws_size,
                              hipStream_t stream) {
  (void)n_in; (void)out_size; (void)ws_size;
  const int B = 8;
  const int N = in_sizes[0] / (3 * B);  // 4096

  char* wsp = (char*)d_ws;
  auto alloc = [&](size_t bytes) -> char* {
    char* p = wsp;
    wsp += (bytes + 255) & ~(size_t)255;
    return p;
  };

  __bf16* h    = (__bf16*)alloc((size_t)B * N * 8 * 2);
  float*  sq   = (float*) alloc((size_t)B * N * 4);
  int*    idx  = (int*)   alloc((size_t)B * N * 16 * 4);
  __bf16* cat  = (__bf16*)alloc((size_t)B * N * 512 * 2);
  __bf16* w1b  = (__bf16*)alloc(64 * 32 * 2);
  __bf16* w2b  = (__bf16*)alloc(128 * 64 * 2);
  __bf16* w3b  = (__bf16*)alloc(64 * 32 * 2);
  __bf16* w4b  = (__bf16*)alloc(128 * 64 * 2);
  __bf16* cw1b = (__bf16*)alloc(128 * 128 * 2);
  __bf16* c2w1b= (__bf16*)alloc(256 * 128 * 2);
  __bf16* cw2b = (__bf16*)alloc(128 * 128 * 2);
  __bf16* c2w2b= (__bf16*)alloc(256 * 128 * 2);
  __bf16* cwb  = (__bf16*)alloc(512 * 512 * 2);
  float* lbn_s = (float*)alloc(8 * 4);    float* lbn_h = (float*)alloc(8 * 4);
  float* bn1_s = (float*)alloc(64 * 4);   float* bn1_h = (float*)alloc(64 * 4);
  float* bn2_s = (float*)alloc(128 * 4);  float* bn2_h = (float*)alloc(128 * 4);
  float* bn3_s = (float*)alloc(64 * 4);   float* bn3_h = (float*)alloc(64 * 4);
  float* bn4_s = (float*)alloc(128 * 4);  float* bn4_h = (float*)alloc(128 * 4);
  float* ab1_s = (float*)alloc(256 * 4);  float* ab1_h = (float*)alloc(256 * 4);
  float* ab2_s = (float*)alloc(256 * 4);  float* ab2_h = (float*)alloc(256 * 4);
  float* bnf_s = (float*)alloc(512 * 4);  float* bnf_h = (float*)alloc(512 * 4);

  auto in = [&](int i) { return (const float*)d_in[i]; };

  // weights -> bf16 (K-padded)
  prep_w_kernel<<<dim3((64 * 32 + 255) / 256), 256, 0, stream>>>(in(6), w1b, 64, 16, 32);
  prep_w_kernel<<<dim3((128 * 64 + 255) / 256), 256, 0, stream>>>(in(11), w2b, 128, 64, 64);
  prep_w_kernel<<<dim3((64 * 32 + 255) / 256), 256, 0, stream>>>(in(16), w3b, 64, 16, 32);
  prep_w_kernel<<<dim3((128 * 64 + 255) / 256), 256, 0, stream>>>(in(21), w4b, 128, 64, 64);
  prep_w_kernel<<<dim3((128 * 128 + 255) / 256), 256, 0, stream>>>(in(26), cw1b, 128, 128, 128);
  prep_w_kernel<<<dim3((256 * 128 + 255) / 256), 256, 0, stream>>>(in(27), c2w1b, 256, 128, 128);
  prep_w_kernel<<<dim3((128 * 128 + 255) / 256), 256, 0, stream>>>(in(32), cw2b, 128, 128, 128);
  prep_w_kernel<<<dim3((256 * 128 + 255) / 256), 256, 0, stream>>>(in(33), c2w2b, 256, 128, 128);
  prep_w_kernel<<<dim3((512 * 512 + 255) / 256), 256, 0, stream>>>(in(38), cwb, 512, 512, 512);

  // BN -> scale/shift
  prep_bn_kernel<<<1, 256, 0, stream>>>(in(2), in(3), in(4), in(5), lbn_s, lbn_h, 8);
  prep_bn_kernel<<<1, 256, 0, stream>>>(in(7), in(8), in(9), in(10), bn1_s, bn1_h, 64);
  prep_bn_kernel<<<1, 256, 0, stream>>>(in(12), in(13), in(14), in(15), bn2_s, bn2_h, 128);
  prep_bn_kernel<<<1, 256, 0, stream>>>(in(17), in(18), in(19), in(20), bn3_s, bn3_h, 64);
  prep_bn_kernel<<<1, 256, 0, stream>>>(in(22), in(23), in(24), in(25), bn4_s, bn4_h, 128);
  prep_bn_kernel<<<1, 256, 0, stream>>>(in(28), in(29), in(30), in(31), ab1_s, ab1_h, 256);
  prep_bn_kernel<<<1, 256, 0, stream>>>(in(34), in(35), in(36), in(37), ab2_s, ab2_h, 256);
  prep_bn_kernel<<<2, 256, 0, stream>>>(in(39), in(40), in(41), in(42), bnf_s, bnf_h, 512);

  // lift 3 -> 8 channels, bf16 + squared norms
  lift_kernel<<<dim3((B * N + 255) / 256), 256, 0, stream>>>(in(0), in(1), lbn_s, lbn_h,
                                                             h, sq, B, N);

  // shared top-16 neighbor indices (top-12 is a prefix)
  topk_kernel<<<dim3(B * (N / 256)), 256, 0, stream>>>(h, sq, idx, N);

  // fused edge-conv branches (dynamic LDS: Fe + F1 + F2 + SB, bf16)
  size_t smem16 = (size_t)(256 * (32 + 64 + 128) + 16 * 128) * 2;  // 118784 B
  size_t smem12 = (size_t)(192 * (32 + 64 + 128) + 16 * 128) * 2;  //  90112 B
  edgeconv_kernel<16><<<dim3(B * (N / 16)), 128, smem16, stream>>>(
      h, idx, w1b, w2b, cw1b, c2w1b, bn1_s, bn1_h, bn2_s, bn2_h, ab1_s, ab1_h,
      cat, N, 0);
  edgeconv_kernel<12><<<dim3(B * (N / 16)), 128, smem12, stream>>>(
      h, idx, w3b, w4b, cw2b, c2w2b, bn3_s, bn3_h, bn4_s, bn4_h, ab2_s, ab2_h,
      cat, N, 256);

  // final 512x512 conv + BN + ReLU -> f32 out[B][512][N]
  final_conv_kernel<<<dim3((B * N) / 64, 512 / 64), 128, 0, stream>>>(
      cat, cwb, bnf_s, bnf_h, (float*)d_out, N);
}
